// MultiHeadAttentionBlock_65712999628858
// MI455X (gfx1250) — compile-verified, matrix operand order checked by reading, not run on hardware
//
#include <hip/hip_runtime.h>
#include <hip/hip_bf16.h>

typedef __bf16 bf16_t;
typedef __attribute__((ext_vector_type(16))) __bf16 v16bf;
typedef __attribute__((ext_vector_type(8)))  __bf16 v8bf;
typedef __attribute__((ext_vector_type(4)))  __bf16 v4bf;
typedef __attribute__((ext_vector_type(8)))  float  v8f;

#define SEQ 1024
#define DMODEL 1024
#define NHEAD 16
#define DHEAD 64
#define ROWS (8 * SEQ)        // B*N = 8192

__device__ __forceinline__ bf16_t f2bf(float f) {
    union { float f; unsigned u; } a; a.f = f;
    unsigned r = a.u + 0x7FFFu + ((a.u >> 16) & 1u);   // RNE
    unsigned short h = (unsigned short)(r >> 16);
    return __builtin_bit_cast(bf16_t, h);
}

// Gather a 16x32 bf16 fragment. CDNA5 16-bit layout: lanes 0-15 hold
// K {0..7,16..23}, lanes 16-31 hold K {8..15,24..31}; row/col = lane&15.
// Caller pre-offsets the pointer by (lane>=16)*8. Two 128-bit loads land
// directly in the fragment register halves via bit_cast (no elementwise moves).
struct frag_pair { v8bf lo, hi; };
__device__ __forceinline__ v16bf gather_frag(const bf16_t* p) {
    frag_pair t;
    t.lo = *(const v8bf*)p;
    t.hi = *(const v8bf*)(p + 16);
    return __builtin_bit_cast(v16bf, t);
}

__device__ __forceinline__ v8f wmma_bf16(v16bf a, v16bf b, v8f c) {
    return __builtin_amdgcn_wmma_f32_16x16x32_bf16(
        false, a, false, b, (short)0, c, false, false);
}

// ---------------- conversion / packing kernels ----------------

__global__ void cvt_f32_bf16_kernel(const float* __restrict__ in, bf16_t* __restrict__ out, int n) {
    int i = (blockIdx.x * 256 + threadIdx.x) * 8;
    if (i >= n) return;
    float4 f0 = *(const float4*)(in + i);
    float4 f1 = *(const float4*)(in + i + 4);
    v8bf o;
    o[0] = f2bf(f0.x); o[1] = f2bf(f0.y); o[2] = f2bf(f0.z); o[3] = f2bf(f0.w);
    o[4] = f2bf(f1.x); o[5] = f2bf(f1.y); o[6] = f2bf(f1.z); o[7] = f2bf(f1.w);
    *(v8bf*)(out + i) = o;
}

// Wq[h][d][e] (fp32) -> transposed bf16 weight Wt[h*64+e][d]  (K contiguous)
__global__ void pack_wqkv_t_kernel(const float* __restrict__ w, bf16_t* __restrict__ out) {
    int i = blockIdx.x * 256 + threadIdx.x;      // i = c*1024 + d
    int c = i >> 10, d = i & 1023;
    int h = c >> 6, e = c & 63;
    out[i] = f2bf(w[((size_t)h * DMODEL + d) * DHEAD + e]);
}

// W[d][j] (fp32, row-major) -> Wt[j][d] bf16
__global__ void pack_wt_kernel(const float* __restrict__ w, bf16_t* __restrict__ out) {
    int i = blockIdx.x * 256 + threadIdx.x;      // i = j*1024 + d
    int j = i >> 10, d = i & 1023;
    out[i] = f2bf(w[(size_t)d * DMODEL + j]);
}

// ---------------- bf16 WMMA GEMM (weights pre-transposed, no LDS) ----------------
// C[M,N] = A[M,K] @ Wt[N,K]^T ; val = (acc + bias[col]) * scale; opt ReLU;
// opt += resid; OUTB_MODE: 0 none, 1 row-major, 2 scatter [B,H,N,Dh],
// 3 scatter transposed [B,H,Dh,N].
template <int OUTB_MODE, bool RELU, bool HAS_RESID, bool HAS_OUTF>
__global__ __launch_bounds__(256)
void gemm_bf16_kernel(const bf16_t* __restrict__ A, const bf16_t* __restrict__ Wt,
                      const float* __restrict__ bias, const float* __restrict__ resid,
                      float* __restrict__ outf, bf16_t* __restrict__ outb,
                      int M, int N, int K, float scale) {
    const int lane = threadIdx.x & 31, wave = threadIdx.x >> 5;
    const int hi = lane >> 4, ln16 = lane & 15;
    const int m0 = blockIdx.y * 128 + (wave & 3) * 32;   // wave: 32 rows x 64 cols
    const int c0 = blockIdx.x * 128 + (wave >> 2) * 64;

    v8f acc[2][4] = {};
    const bf16_t* arow0 = A + (size_t)(m0 + ln16) * K + hi * 8;
    const bf16_t* arow1 = arow0 + (size_t)16 * K;
    const bf16_t* brow  = Wt + (size_t)(c0 + ln16) * K + hi * 8;

#pragma unroll 2
    for (int k0 = 0; k0 < K; k0 += 32) {
        __builtin_prefetch(arow0 + k0 + 64, 0, 0);       // speculative, 2 chunks ahead
        __builtin_prefetch(brow + k0 + 64, 0, 0);
        v16bf a0 = gather_frag(arow0 + k0);
        v16bf a1 = gather_frag(arow1 + k0);
#pragma unroll
        for (int t = 0; t < 4; ++t) {
            v16bf bf = gather_frag(brow + (size_t)t * 16 * K + k0);
            acc[0][t] = wmma_bf16(a0, bf, acc[0][t]);
            acc[1][t] = wmma_bf16(a1, bf, acc[1][t]);
        }
    }
#pragma unroll
    for (int rg = 0; rg < 2; ++rg)
#pragma unroll
        for (int t = 0; t < 4; ++t) {
            int col = c0 + t * 16 + ln16;
            float bv = bias[col];
#pragma unroll
            for (int r = 0; r < 8; ++r) {
                int row = m0 + rg * 16 + r + 8 * hi;
                float v = (acc[rg][t][r] + bv) * scale;
                if (RELU) v = fmaxf(v, 0.f);
                if (HAS_RESID) v += resid[(size_t)row * N + col];
                if (HAS_OUTF) outf[(size_t)row * N + col] = v;
                if (OUTB_MODE == 1) {
                    outb[(size_t)row * N + col] = f2bf(v);
                } else if (OUTB_MODE == 2) {           // [B,H,N,Dh]
                    int b = row >> 10, n = row & 1023;
                    int h = col >> 6, e = col & 63;
                    outb[(((size_t)b * NHEAD + h) * SEQ + n) * DHEAD + e] = f2bf(v);
                } else if (OUTB_MODE == 3) {           // [B,H,Dh,N] (transposed V)
                    int b = row >> 10, n = row & 1023;
                    int h = col >> 6, e = col & 63;
                    outb[(((size_t)b * NHEAD + h) * DHEAD + e) * SEQ + n] = f2bf(v);
                }
            }
        }
}

// ---------------- fused flash attention (64-key chunks) ----------------
// Q pre-scaled by 1/sqrt(H*Dh); Q,K in [B,H,N,Dh]; V in [B,H,Dh,N] (transposed)
// so K and V fragments are direct global b128 gathers — no barriers, no staging.
// Output written with interleaved head concat: out[b, n, e*H + h].
__global__ __launch_bounds__(256)
void attention_kernel(const bf16_t* __restrict__ Q, const bf16_t* __restrict__ Kb,
                      const bf16_t* __restrict__ Vt, float* __restrict__ out) {
    __shared__ bf16_t pls[8][16 * 64];            // per-wave p tile (row-major 16x64)
    const int tid = threadIdx.x, lane = tid & 31, wave = tid >> 5;
    const int hi = lane >> 4, ln16 = lane & 15;
    const int b = blockIdx.z, h = blockIdx.y;
    const int q0 = (blockIdx.x * 8 + wave) * 16;
    const size_t ho = ((size_t)b * NHEAD + h) * SEQ * DHEAD;
    const bf16_t* qh = Q + ho;
    const bf16_t* kh = Kb + ho;
    const bf16_t* vh = Vt + ho;                    // [Dh][SEQ]

    v16bf qf[2];
#pragma unroll
    for (int ec = 0; ec < 2; ++ec)
        qf[ec] = gather_frag(qh + (size_t)(q0 + ln16) * DHEAD + ec * 32 + hi * 8);

    float mrow[8], lrow[8];
    v8f o[4] = {};
#pragma unroll
    for (int r = 0; r < 8; ++r) { mrow[r] = -3.0e38f; lrow[r] = 0.f; }

    for (int k0 = 0; k0 < SEQ; k0 += 64) {
        // scores for 64 keys: 4 tiles x (K=64 -> 2 WMMA)
        v8f s[4] = {};
#pragma unroll
        for (int kt = 0; kt < 4; ++kt)
#pragma unroll
            for (int ec = 0; ec < 2; ++ec) {
                v16bf kf = gather_frag(kh + (size_t)(k0 + kt * 16 + ln16) * DHEAD + ec * 32 + hi * 8);
                s[kt] = wmma_bf16(qf[ec], kf, s[kt]);
            }

        float alpha[8];
#pragma unroll
        for (int r = 0; r < 8; ++r) {             // online softmax, row = r + 8*hi
            float mx = fmaxf(fmaxf(s[0][r], s[1][r]), fmaxf(s[2][r], s[3][r]));
#pragma unroll
            for (int off = 8; off >= 1; off >>= 1) mx = fmaxf(mx, __shfl_xor(mx, off, 16));
            float mn = fmaxf(mrow[r], mx);
            alpha[r] = __expf(mrow[r] - mn);
            mrow[r] = mn;
            float p0 = __expf(s[0][r] - mn), p1 = __expf(s[1][r] - mn);
            float p2 = __expf(s[2][r] - mn), p3 = __expf(s[3][r] - mn);
            float rs = (p0 + p1) + (p2 + p3);
#pragma unroll
            for (int off = 8; off >= 1; off >>= 1) rs += __shfl_xor(rs, off, 16);
            lrow[r] = lrow[r] * alpha[r] + rs;
            int row = r + 8 * hi;
            pls[wave][row * 64 + ln16]      = f2bf(p0);
            pls[wave][row * 64 + 16 + ln16] = f2bf(p1);
            pls[wave][row * 64 + 32 + ln16] = f2bf(p2);
            pls[wave][row * 64 + 48 + ln16] = f2bf(p3);
        }
#pragma unroll
        for (int t = 0; t < 4; ++t)
#pragma unroll
            for (int r = 0; r < 8; ++r) o[t][r] *= alpha[r];

        // PV: 2 p-fragments (keys 0-31, 32-63) x 4 V column tiles
#pragma unroll
        for (int c = 0; c < 2; ++c) {
            v16bf pf = gather_frag(&pls[wave][ln16 * 64 + c * 32 + hi * 8]);
#pragma unroll
            for (int t = 0; t < 4; ++t) {
                v16bf vf = gather_frag(vh + (size_t)(t * 16 + ln16) * SEQ + k0 + c * 32 + hi * 8);
                o[t] = wmma_bf16(pf, vf, o[t]);
            }
        }
    }
#pragma unroll
    for (int t = 0; t < 4; ++t) {
        int e = t * 16 + ln16;
#pragma unroll
        for (int r = 0; r < 8; ++r) {
            int n = q0 + r + 8 * hi;
            out[((size_t)b * SEQ + n) * DMODEL + e * NHEAD + h] = o[t][r] / lrow[r];
        }
    }
}

// ---------------- layernorm (+optional residual input, bf16 side output) ----------------
__global__ __launch_bounds__(256)
void ln_kernel(const float* __restrict__ a, const float* __restrict__ b,
               const float* __restrict__ g, const float* __restrict__ beta,
               float* __restrict__ outf, bf16_t* __restrict__ outb) {
    const int row = blockIdx.x, tid = threadIdx.x;
    __shared__ float red[2][8];
    const size_t base = (size_t)row * DMODEL + tid * 4;
    float4 xv = *(const float4*)(a + base);
    if (b) {
        float4 rv = *(const float4*)(b + base);
        xv.x += rv.x; xv.y += rv.y; xv.z += rv.z; xv.w += rv.w;
    }
    float s = xv.x + xv.y + xv.z + xv.w;
    float sq = xv.x * xv.x + xv.y * xv.y + xv.z * xv.z + xv.w * xv.w;
#pragma unroll
    for (int off = 16; off >= 1; off >>= 1) {
        s += __shfl_xor(s, off, 32); sq += __shfl_xor(sq, off, 32);
    }
    if ((tid & 31) == 0) { red[0][tid >> 5] = s; red[1][tid >> 5] = sq; }
    __syncthreads();
    s = 0.f; sq = 0.f;
#pragma unroll
    for (int w = 0; w < 8; ++w) { s += red[0][w]; sq += red[1][w]; }
    float mu = s * (1.f / DMODEL);
    float var = sq * (1.f / DMODEL) - mu * mu;
    float rstd = rsqrtf(var + 1e-5f);
    float4 gv = *(const float4*)(g + tid * 4);
    float4 bv = *(const float4*)(beta + tid * 4);
    float y0 = (xv.x - mu) * rstd * gv.x + bv.x;
    float y1 = (xv.y - mu) * rstd * gv.y + bv.y;
    float y2 = (xv.z - mu) * rstd * gv.z + bv.z;
    float y3 = (xv.w - mu) * rstd * gv.w + bv.w;
    if (outf) { float4 o; o.x = y0; o.y = y1; o.z = y2; o.w = y3; *(float4*)(outf + base) = o; }
    if (outb) {
        v4bf o; o[0] = f2bf(y0); o[1] = f2bf(y1); o[2] = f2bf(y2); o[3] = f2bf(y3);
        *(v4bf*)(outb + base) = o;
    }
}

// ---------------- host orchestration ----------------
extern "C" void kernel_launch(void* const* d_in, const int* in_sizes, int n_in,
                              void* d_out, int out_size, void* d_ws, size_t ws_size,
                              hipStream_t stream) {
    (void)in_sizes; (void)n_in; (void)out_size; (void)ws_size;
    const float* queries = (const float*)d_in[0];
    const float* keysI   = (const float*)d_in[1];
    const float* valuesI = (const float*)d_in[2];
    const float* Wq = (const float*)d_in[3];
    const float* bq = (const float*)d_in[4];
    const float* Wk = (const float*)d_in[5];
    const float* bk = (const float*)d_in[6];
    const float* Wv = (const float*)d_in[7];
    const float* bv = (const float*)d_in[8];
    const float* W1 = (const float*)d_in[9];
    const float* b1 = (const float*)d_in[10];
    const float* W2 = (const float*)d_in[11];
    const float* b2 = (const float*)d_in[12];
    const float* g1  = (const float*)d_in[13];
    const float* be1 = (const float*)d_in[14];
    const float* g2  = (const float*)d_in[15];
    const float* be2 = (const float*)d_in[16];

    char* ws = (char*)d_ws;
    const size_t MB = 1ull << 20;
    bf16_t* xq = (bf16_t*)(ws + 0);          // 16 MB each
    bf16_t* xk = (bf16_t*)(ws + 16 * MB);
    bf16_t* xv = (bf16_t*)(ws + 32 * MB);
    bf16_t* wq = (bf16_t*)(ws + 48 * MB);    // 2 MB each (transposed bf16)
    bf16_t* wk = (bf16_t*)(ws + 50 * MB);
    bf16_t* wv = (bf16_t*)(ws + 52 * MB);
    bf16_t* w1 = (bf16_t*)(ws + 54 * MB);
    bf16_t* w2 = (bf16_t*)(ws + 56 * MB);
    bf16_t* qb = (bf16_t*)(ws + 58 * MB);    // 16 MB [B,H,N,Dh]
    bf16_t* kb = (bf16_t*)(ws + 74 * MB);    // 16 MB [B,H,N,Dh]
    bf16_t* vb = (bf16_t*)(ws + 90 * MB);    // 16 MB [B,H,Dh,N] (transposed)
    float*  mha = (float*)(ws + 0);          // 32 MB, reuses xq+xk (dead after QKV)
    float*  h1f = (float*)(ws + 106 * MB);   // 32 MB
    bf16_t* h1b = (bf16_t*)(ws + 32 * MB);   // 16 MB, reuses xv
    bf16_t* f1b = (bf16_t*)(ws + 58 * MB);   // 16 MB, reuses qb (dead after attention)
    float*  x2f = (float*)(ws + 74 * MB);    // 32 MB, reuses kb+vb

    const int NACT = ROWS * DMODEL;          // 8,388,608
    const int NW   = DMODEL * DMODEL;        // 1,048,576
    const float qscale = 1.0f / 32.0f;       // 1/sqrt(H*Dh)

    // precision conversion / weight packing (weights stored transposed)
    cvt_f32_bf16_kernel<<<NACT / 2048, 256, 0, stream>>>(queries, xq, NACT);
    cvt_f32_bf16_kernel<<<NACT / 2048, 256, 0, stream>>>(keysI,   xk, NACT);
    cvt_f32_bf16_kernel<<<NACT / 2048, 256, 0, stream>>>(valuesI, xv, NACT);
    pack_wqkv_t_kernel<<<NW / 256, 256, 0, stream>>>(Wq, wq);
    pack_wqkv_t_kernel<<<NW / 256, 256, 0, stream>>>(Wk, wk);
    pack_wqkv_t_kernel<<<NW / 256, 256, 0, stream>>>(Wv, wv);
    pack_wt_kernel<<<NW / 256, 256, 0, stream>>>(W1, w1);
    pack_wt_kernel<<<NW / 256, 256, 0, stream>>>(W2, w2);

    // QKV projections (WMMA): Q pre-scaled; V scattered transposed
    dim3 ggrid(DMODEL / 128, ROWS / 128);
    gemm_bf16_kernel<2, false, false, false><<<ggrid, 256, 0, stream>>>(
        xq, wq, bq, nullptr, nullptr, qb, ROWS, DMODEL, DMODEL, qscale);
    gemm_bf16_kernel<2, false, false, false><<<ggrid, 256, 0, stream>>>(
        xk, wk, bk, nullptr, nullptr, kb, ROWS, DMODEL, DMODEL, 1.0f);
    gemm_bf16_kernel<3, false, false, false><<<ggrid, 256, 0, stream>>>(
        xv, wv, bv, nullptr, nullptr, vb, ROWS, DMODEL, DMODEL, 1.0f);

    // fused flash attention + interleaved head concat
    attention_kernel<<<dim3(SEQ / 128, NHEAD, 8), 256, 0, stream>>>(qb, kb, vb, mha);

    // residual + LN1 -> h1 (fp32 + bf16)
    ln_kernel<<<ROWS, 256, 0, stream>>>(mha, queries, g1, be1, h1f, h1b);

    // FFN1: relu(h1 @ W1 + b1) -> bf16
    gemm_bf16_kernel<1, true, false, false><<<ggrid, 256, 0, stream>>>(
        h1b, w1, b1, nullptr, nullptr, f1b, ROWS, DMODEL, DMODEL, 1.0f);
    // FFN2: h1 + (f1 @ W2 + b2) -> fp32
    gemm_bf16_kernel<0, false, true, true><<<ggrid, 256, 0, stream>>>(
        f1b, w2, b2, h1f, x2f, nullptr, ROWS, DMODEL, DMODEL, 1.0f);
    // final LN -> output
    ln_kernel<<<ROWS, 256, 0, stream>>>(x2f, nullptr, g2, be2, (float*)d_out, nullptr);
}
